// Cross_Self_attention_35699768164372
// MI455X (gfx1250) — matrix operand order repaired and checked by reading, not compiled, not run on hardware
//
#include <hip/hip_runtime.h>

// ---------------------------------------------------------------------------
// Cross/Self attention, MI455X (gfx1250, wave32, WMMA).
// Pipeline (all bf16 WMMA w/ f32 accum):
//   1) x,y f32 -> bf16 ; Wq/Wk/Wv f32 -> bf16 transposed [H,D] (LDS transpose)
//   2) q = x*Wq+bq, k = x*Wk+bk, v = y*Wv+bv   (NT-GEMM, bf16 out)
//   3) g  = q .* v, written transposed [B,H,S] bf16 (LDS transpose)
//   4) scores = q * k^T (f32, workspace)
//   5) probs = softmax(scores) -> bf16
//   6) out = probs * g   (NT-GEMM vs g^T, f32 out)
// Workspace need ~240 MB (with buffer lifetime overlap).
// ---------------------------------------------------------------------------

typedef __bf16 bf16_t;
typedef __attribute__((ext_vector_type(16))) __bf16 v16bf;
typedef __attribute__((ext_vector_type(8)))  __bf16 v8bf;
typedef __attribute__((ext_vector_type(8)))  float  v8f;

static __device__ __forceinline__ unsigned short f2bf(float f) {
    unsigned int u = __float_as_uint(f);
    u = (u + 0x7FFFu + ((u >> 16) & 1u)) >> 16;   // round-to-nearest-even
    return (unsigned short)u;
}
static __device__ __forceinline__ float bf2f(unsigned short h) {
    return __uint_as_float(((unsigned int)h) << 16);
}

// ---------------------------------------------------------------------------
// f32 -> bf16 flat convert (4 elements / thread iteration, packed 8B stores)
// ---------------------------------------------------------------------------
__global__ void f32_to_bf16_kernel(const float4* __restrict__ in,
                                   unsigned long long* __restrict__ out, long n4) {
    long i = (long)blockIdx.x * blockDim.x + threadIdx.x;
    long stride = (long)gridDim.x * blockDim.x;
    for (; i < n4; i += stride) {
        float4 f = in[i];
        unsigned long long p = (unsigned long long)f2bf(f.x)
                             | ((unsigned long long)f2bf(f.y) << 16)
                             | ((unsigned long long)f2bf(f.z) << 32)
                             | ((unsigned long long)f2bf(f.w) << 48);
        out[i] = p;
    }
}

// ---------------------------------------------------------------------------
// W[D,H] f32 -> Wt[H,D] bf16, 32x32 LDS tile transpose. block (32,8)
// ---------------------------------------------------------------------------
__global__ __launch_bounds__(256)
void transpose_w_kernel(const float* __restrict__ W, unsigned short* __restrict__ Wt,
                        int D, int H) {
    __shared__ unsigned short tile[32][33];
    const int h0 = blockIdx.x * 32, d0 = blockIdx.y * 32;
    const int tx = threadIdx.x, ty = threadIdx.y;
#pragma unroll
    for (int i = 0; i < 4; ++i)
        tile[ty + i * 8][tx] = f2bf(W[(long)(d0 + ty + i * 8) * H + h0 + tx]);
    __syncthreads();
#pragma unroll
    for (int i = 0; i < 4; ++i)
        Wt[(long)(h0 + ty + i * 8) * D + d0 + tx] = tile[tx][ty + i * 8];
}

// ---------------------------------------------------------------------------
// g = q .* v (bf16 in), written TRANSPOSED: gt[b, h, s]. block (32,8)
// ---------------------------------------------------------------------------
__global__ __launch_bounds__(256)
void gate_transpose_kernel(const unsigned short* __restrict__ q,
                           const unsigned short* __restrict__ v,
                           unsigned short* __restrict__ gt, int S, int H) {
    __shared__ unsigned short tile[32][33];
    const long base  = (long)blockIdx.z * S * H;
    const long baseT = (long)blockIdx.z * H * S;
    const int h0 = blockIdx.x * 32, s0 = blockIdx.y * 32;
    const int tx = threadIdx.x, ty = threadIdx.y;
#pragma unroll
    for (int i = 0; i < 4; ++i) {
        long idx = base + (long)(s0 + ty + i * 8) * H + h0 + tx;
        tile[ty + i * 8][tx] = f2bf(bf2f(q[idx]) * bf2f(v[idx]));
    }
    __syncthreads();
#pragma unroll
    for (int i = 0; i < 4; ++i)
        gt[baseT + (long)(h0 + ty + i * 8) * S + s0 + tx] = tile[tx][ty + i * 8];
}

// ---------------------------------------------------------------------------
// Row softmax over S=2048 f32 scores -> bf16 probs. One block (256 thr) / row.
// ---------------------------------------------------------------------------
__global__ __launch_bounds__(256)
void softmax_row_kernel(const float* __restrict__ scores,
                        unsigned short* __restrict__ probs, int S) {
    __shared__ float red[8];
    const long row = blockIdx.x;
    const float* src = scores + row * (long)S;
    unsigned short* dst = probs + row * (long)S;
    const int t = threadIdx.x, lane = t & 31, wave = t >> 5;

    float v[8];
    float m = -3.402823466e38f;
#pragma unroll
    for (int i = 0; i < 8; ++i) { v[i] = src[t + 256 * i]; m = fmaxf(m, v[i]); }
#pragma unroll
    for (int o = 16; o > 0; o >>= 1) m = fmaxf(m, __shfl_xor(m, o, 32));
    if (lane == 0) red[wave] = m;
    __syncthreads();
#pragma unroll
    for (int j = 0; j < 8; ++j) m = fmaxf(m, red[j]);
    __syncthreads();

    float s = 0.f;
#pragma unroll
    for (int i = 0; i < 8; ++i) { v[i] = __expf(v[i] - m); s += v[i]; }
#pragma unroll
    for (int o = 16; o > 0; o >>= 1) s += __shfl_xor(s, o, 32);
    if (lane == 0) red[wave] = s;
    __syncthreads();
    s = 0.f;
#pragma unroll
    for (int j = 0; j < 8; ++j) s += red[j];
    const float inv = 1.0f / s;
#pragma unroll
    for (int i = 0; i < 8; ++i) dst[t + 256 * i] = f2bf(v[i] * inv);
}

// ---------------------------------------------------------------------------
// NT-GEMM: C[M,N] = A[M,K] * Bt[N,K]^T (+bias[n]).  bf16 in, f32 accum.
// Block: 128 thr = 4 waves (2x2) -> 64x128 C tile. Wave: 32x64 (2x4 frags).
// Per-lane fragment loads (A and B share one contiguous k order, not AMD's table order):
//   A (16x32 bf16): lane holds row (lane&15), 16 contiguous k at half*16
//   B (32x16 bf16): lane holds col (lane&15) = row of Bt, 16 contiguous k at half*16
//   C (16x16 f32):  vgpr r -> (m = r + half*8, n = lane&15)
// ---------------------------------------------------------------------------
template <bool BF16OUT, bool BIAS>
__global__ __launch_bounds__(128)
void gemm_nt_kernel(const bf16_t* __restrict__ A, const bf16_t* __restrict__ Bt,
                    const float* __restrict__ bias, void* __restrict__ Cout,
                    int M, int N, int K, long batchA, long batchB, long batchC) {
    const int lane = threadIdx.x & 31;
    const int wave = threadIdx.x >> 5;
    const int half = lane >> 4;
    const int l16  = lane & 15;
    const long z   = blockIdx.z;

    A  += z * batchA;
    Bt += z * batchB;

    const int m_wave = blockIdx.x * 64  + (wave >> 1) * 32;
    const int n_wave = blockIdx.y * 128 + (wave & 1) * 64;

    const bf16_t* arow0 = A + (long)(m_wave + l16) * K + half * 16;
    const bf16_t* arow1 = arow0 + (long)16 * K;
    const bf16_t* brow0 = Bt + (long)(n_wave +  0 + l16) * K + half * 16;
    const bf16_t* brow1 = Bt + (long)(n_wave + 16 + l16) * K + half * 16;
    const bf16_t* brow2 = Bt + (long)(n_wave + 32 + l16) * K + half * 16;
    const bf16_t* brow3 = Bt + (long)(n_wave + 48 + l16) * K + half * 16;

    v8f acc[2][4] = {};

    for (int k = 0; k < K; k += 32) {
        if (k + 128 < K) {  // global_prefetch_b8 of upcoming k-tiles
            __builtin_prefetch(arow0 + k + 128, 0, 1);
            __builtin_prefetch(arow1 + k + 128, 0, 1);
            __builtin_prefetch(brow0 + k + 128, 0, 1);
            __builtin_prefetch(brow2 + k + 128, 0, 1);
        }
        v16bf a0 = *(const v16bf*)(arow0 + k);
        v16bf a1 = *(const v16bf*)(arow1 + k);
        v16bf b0 = *(const v16bf*)(brow0 + k);
        v16bf b1 = *(const v16bf*)(brow1 + k);
        v16bf b2 = *(const v16bf*)(brow2 + k);
        v16bf b3 = *(const v16bf*)(brow3 + k);

        acc[0][0] = __builtin_amdgcn_wmma_f32_16x16x32_bf16(false, a0, false, b0, (short)0, acc[0][0], false, false);
        acc[0][1] = __builtin_amdgcn_wmma_f32_16x16x32_bf16(false, a0, false, b1, (short)0, acc[0][1], false, false);
        acc[0][2] = __builtin_amdgcn_wmma_f32_16x16x32_bf16(false, a0, false, b2, (short)0, acc[0][2], false, false);
        acc[0][3] = __builtin_amdgcn_wmma_f32_16x16x32_bf16(false, a0, false, b3, (short)0, acc[0][3], false, false);
        acc[1][0] = __builtin_amdgcn_wmma_f32_16x16x32_bf16(false, a1, false, b0, (short)0, acc[1][0], false, false);
        acc[1][1] = __builtin_amdgcn_wmma_f32_16x16x32_bf16(false, a1, false, b1, (short)0, acc[1][1], false, false);
        acc[1][2] = __builtin_amdgcn_wmma_f32_16x16x32_bf16(false, a1, false, b2, (short)0, acc[1][2], false, false);
        acc[1][3] = __builtin_amdgcn_wmma_f32_16x16x32_bf16(false, a1, false, b3, (short)0, acc[1][3], false, false);
    }

#pragma unroll
    for (int ni = 0; ni < 4; ++ni) {
        const int col = n_wave + ni * 16 + l16;
        const float bv = BIAS ? bias[col] : 0.0f;
#pragma unroll
        for (int mi = 0; mi < 2; ++mi) {
            const int rbase = m_wave + mi * 16 + half * 8;
#pragma unroll
            for (int r = 0; r < 8; ++r) {
                const float val = acc[mi][ni][r] + bv;
                const long idx = z * batchC + (long)(rbase + r) * N + col;
                if (BF16OUT) ((unsigned short*)Cout)[idx] = f2bf(val);
                else         ((float*)Cout)[idx] = val;
            }
        }
    }
}

// ---------------------------------------------------------------------------
extern "C" void kernel_launch(void* const* d_in, const int* in_sizes, int n_in,
                              void* d_out, int out_size, void* d_ws, size_t ws_size,
                              hipStream_t stream) {
    constexpr int  B = 8, S = 2048, D = 768, H = 768;
    constexpr long BSD = (long)B * S * D;          // 12.58M elements
    constexpr long BSH = (long)B * S * H;

    const float* x  = (const float*)d_in[0];
    const float* y  = (const float*)d_in[1];
    const float* Wq = (const float*)d_in[2];
    const float* bq = (const float*)d_in[3];
    const float* Wk = (const float*)d_in[4];
    const float* bk = (const float*)d_in[5];
    const float* Wv = (const float*)d_in[6];
    const float* bv = (const float*)d_in[7];
    float* out = (float*)d_out;

    // ---- workspace carve-up (aligned 256B) -------------------------------
    char*  ws  = (char*)d_ws;
    size_t off = 0;
    auto alloc = [&](size_t bytes) -> void* {
        off = (off + 255) & ~(size_t)255;
        void* p = ws + off;
        off += bytes;
        return p;
    };
    unsigned short* wtq = (unsigned short*)alloc((size_t)D * H * 2);
    unsigned short* wtk = (unsigned short*)alloc((size_t)D * H * 2);
    unsigned short* wtv = (unsigned short*)alloc((size_t)D * H * 2);
    unsigned short* qbf = (unsigned short*)alloc((size_t)BSH * 2);
    unsigned short* kbf = (unsigned short*)alloc((size_t)BSH * 2);
    unsigned short* vbf = (unsigned short*)alloc((size_t)BSH * 2);
    unsigned short* gt  = (unsigned short*)alloc((size_t)BSH * 2);   // [B,H,S]
    // region X: x_bf + y_bf early; scores f32 later (scores superset)
    char* regX = (char*)alloc((size_t)B * S * S * 4);                // 134 MB
    unsigned short* xbf = (unsigned short*)regX;
    unsigned short* ybf = (unsigned short*)(regX + (size_t)BSD * 2);
    float* scores = (float*)regX;                                    // lives after x/y dead
    unsigned short* probs = qbf;  // 67MB overlay on q/k/(part of v): all dead by then

    // ---- 1) conversions ---------------------------------------------------
    f32_to_bf16_kernel<<<2048, 256, 0, stream>>>((const float4*)x, (unsigned long long*)xbf, BSD / 4);
    f32_to_bf16_kernel<<<2048, 256, 0, stream>>>((const float4*)y, (unsigned long long*)ybf, BSD / 4);
    dim3 t328(32, 8);
    dim3 gW(H / 32, D / 32);
    transpose_w_kernel<<<gW, t328, 0, stream>>>(Wq, wtq, D, H);
    transpose_w_kernel<<<gW, t328, 0, stream>>>(Wk, wtk, D, H);
    transpose_w_kernel<<<gW, t328, 0, stream>>>(Wv, wtv, D, H);

    // ---- 2) projections: [B*S,D] x [H,D]^T -> bf16 [B*S,H] ----------------
    dim3 g1((B * S) / 64, H / 128, 1);
    gemm_nt_kernel<true, true><<<g1, 128, 0, stream>>>(
        (const bf16_t*)xbf, (const bf16_t*)wtq, bq, qbf, B * S, H, D, 0, 0, 0);
    gemm_nt_kernel<true, true><<<g1, 128, 0, stream>>>(
        (const bf16_t*)xbf, (const bf16_t*)wtk, bk, kbf, B * S, H, D, 0, 0, 0);
    gemm_nt_kernel<true, true><<<g1, 128, 0, stream>>>(
        (const bf16_t*)ybf, (const bf16_t*)wtv, bv, vbf, B * S, H, D, 0, 0, 0);

    // ---- 3) gate g = q.*v, transposed to [B,H,S] --------------------------
    gate_transpose_kernel<<<dim3(H / 32, S / 32, B), t328, 0, stream>>>(qbf, vbf, gt, S, H);

    // ---- 4) scores = q k^T  (per batch 2048x2048, K=768) ------------------
    dim3 g2(S / 64, S / 128, B);
    gemm_nt_kernel<false, false><<<g2, 128, 0, stream>>>(
        (const bf16_t*)qbf, (const bf16_t*)kbf, nullptr, scores,
        S, S, H, (long)S * H, (long)S * H, (long)S * S);

    // ---- 5) softmax -> bf16 probs ----------------------------------------
    softmax_row_kernel<<<B * S, 256, 0, stream>>>(scores, probs, S);

    // ---- 6) out = probs * g  (A=[S,S] probs, Bt=[H,S] g^T) ----------------
    dim3 g3(S / 64, H / 128, B);
    gemm_nt_kernel<false, false><<<g3, 128, 0, stream>>>(
        (const bf16_t*)probs, (const bf16_t*)gt, nullptr, out,
        S, H, S, (long)S * S, (long)H * S, (long)S * H);
}
